// ChunkProcessor_111669150375
// MI455X (gfx1250) — compile-verified, matrix operand order checked by reading, not run on hardware
//
#include <hip/hip_runtime.h>
#include <hip/hip_bf16.h>

#define Bdim 8
#define Sdim 8192
#define Ddim 512
#define Cdim 512
#define Hn 8
#define HD 64
#define Lnum 2
#define DFFdim 2048

typedef __attribute__((ext_vector_type(16))) _Float16 v16h;
typedef __attribute__((ext_vector_type(8)))  _Float16 v8h;
typedef __attribute__((ext_vector_type(4)))  _Float16 v4h;
typedef __attribute__((ext_vector_type(8)))  float    v8f;

__device__ __forceinline__ v8f wmma32(v16h a, v16h b, v8f c) {
  return __builtin_amdgcn_wmma_f32_16x16x32_f16(
      /*neg_a=*/false, a, /*neg_b=*/false, b,
      /*c_mod=*/(short)0, c, /*reuse_a=*/false, /*reuse_b=*/false);
}

// Fragment load: elements 0..7 from p[0..7], 8..15 from p[16..23] (16B each).
__device__ __forceinline__ v16h load_frag(const _Float16* p) {
  v16h v;
  *(v8h*)&v       = *(const v8h*)p;
  *((v8h*)&v + 1) = *(const v8h*)(p + 16);
  return v;
}

// ---------------------------------------------------------------------------
// f32 -> f16 cast (n multiple of 4)
// ---------------------------------------------------------------------------
__global__ __launch_bounds__(256) void cast_f16(const float* __restrict__ src,
                                                _Float16* __restrict__ dst, int n) {
  int i = blockIdx.x * 256 + threadIdx.x;
  if (i * 4 >= n) return;
  float4 v = ((const float4*)src)[i];
  v4h o;
  o[0] = (_Float16)v.x; o[1] = (_Float16)v.y;
  o[2] = (_Float16)v.z; o[3] = (_Float16)v.w;
  ((v4h*)dst)[i] = o;
}

// ---------------------------------------------------------------------------
// Segment boundaries: start[b][c] = first token index of chunk c; start[b][C]=S
// ---------------------------------------------------------------------------
__global__ __launch_bounds__(256) void seg_bounds(const int* __restrict__ seg,
                                                  int* __restrict__ start) {
  int i = blockIdx.x * 256 + threadIdx.x;          // over B*S
  int b = i / Sdim, s = i - b * Sdim;
  int* sb = start + b * (Cdim + 1);
  if (s == 0) {
    sb[0] = 0;
    sb[Cdim] = Sdim;
  } else {
    int cur = seg[i], prev = seg[i - 1];
    for (int c = prev + 1; c <= cur; ++c) sb[c] = s;   // ids sorted
  }
}

// ---------------------------------------------------------------------------
// Per-chunk mean pooling -> x (f32) and xh (f16)
// ---------------------------------------------------------------------------
__global__ __launch_bounds__(128) void seg_pool(const float* __restrict__ tokens,
                                                const int* __restrict__ start,
                                                float* __restrict__ X,
                                                _Float16* __restrict__ Xh) {
  int row = blockIdx.x;                // b*C + c
  int b = row >> 9, c = row & (Cdim - 1);
  const int* sb = start + b * (Cdim + 1);
  int s0 = sb[c], s1 = sb[c + 1];
  const int t = threadIdx.x;
  float a0 = 0.f, a1 = 0.f, a2 = 0.f, a3 = 0.f;
  for (int s = s0; s < s1; ++s) {
    const float* tp = tokens + ((size_t)b * Sdim + s) * Ddim;
    a0 += tp[t]; a1 += tp[t + 128]; a2 += tp[t + 256]; a3 += tp[t + 384];
  }
  float inv = 1.f / (float)(s1 - s0);
  a0 *= inv; a1 *= inv; a2 *= inv; a3 *= inv;
  float* xp = X + (size_t)row * Ddim;
  xp[t] = a0; xp[t + 128] = a1; xp[t + 256] = a2; xp[t + 384] = a3;
  _Float16* hp = Xh + (size_t)row * Ddim;
  hp[t] = (_Float16)a0; hp[t + 128] = (_Float16)a1;
  hp[t + 256] = (_Float16)a2; hp[t + 384] = (_Float16)a3;
}

// ---------------------------------------------------------------------------
// GEMM: Out[M,N] = A[M,K](f16) * W[N,K](f16)^T + bias, opt ReLU,
// f32 or f16 output. One wave per 32x64 output tile (2x4 WMMA accumulators).
// ---------------------------------------------------------------------------
__global__ __launch_bounds__(256) void gemm_wmma(const _Float16* __restrict__ A,
                                                 const _Float16* __restrict__ W,
                                                 const float* __restrict__ bias,
                                                 void* __restrict__ Out,
                                                 int M, int N, int K,
                                                 int doRelu, int outHalf) {
  const int wid = blockIdx.x * 8 + (threadIdx.x >> 5);
  const int ntiles = N >> 6;
  const int tm = wid / ntiles;
  const int tn = wid - tm * ntiles;
  if (tm * 32 >= M) return;
  const int lane = threadIdx.x & 31;
  const int r = lane & 15;
  const int koff = (lane & 16) ? 8 : 0;
  const _Float16* a0p = A + (size_t)(tm * 32 + r) * K + koff;
  const _Float16* a1p = a0p + (size_t)16 * K;
  const _Float16* bp0 = W + (size_t)(tn * 64 + r) * K + koff;
  v8f acc[2][4] = {};
  for (int k = 0; k < K; k += 32) {
    v16h a0 = load_frag(a0p + k);
    v16h a1 = load_frag(a1p + k);
#pragma unroll
    for (int nn = 0; nn < 4; ++nn) {
      v16h bf = load_frag(bp0 + (size_t)(nn * 16) * K + k);
      acc[0][nn] = wmma32(a0, bf, acc[0][nn]);
      acc[1][nn] = wmma32(a1, bf, acc[1][nn]);
    }
  }
  const int half = lane >> 4;
  const int rowBase = tm * 32 + half * 8;
#pragma unroll
  for (int nn = 0; nn < 4; ++nn) {
    const int col = tn * 64 + nn * 16 + r;
    const float bi = bias[col];
#pragma unroll
    for (int j = 0; j < 8; ++j) {
      float v0 = acc[0][nn][j] + bi;
      float v1 = acc[1][nn][j] + bi;
      if (doRelu) { v0 = fmaxf(v0, 0.f); v1 = fmaxf(v1, 0.f); }
      if (outHalf) {
        ((_Float16*)Out)[(size_t)(rowBase + j) * N + col]      = (_Float16)v0;
        ((_Float16*)Out)[(size_t)(rowBase + 16 + j) * N + col] = (_Float16)v1;
      } else {
        ((float*)Out)[(size_t)(rowBase + j) * N + col]      = v0;
        ((float*)Out)[(size_t)(rowBase + 16 + j) * N + col] = v1;
      }
    }
  }
}

// ---------------------------------------------------------------------------
// Attention: one wave per (b, h, 16-query block). qkv is f16 [B*C, 3D].
// scores via WMMA -> f32 LDS softmax -> f16 probs in LDS -> P*V via WMMA.
// Output written as f16.
// ---------------------------------------------------------------------------
__global__ __launch_bounds__(32) void attn_wmma(const _Float16* __restrict__ qkv,
                                                _Float16* __restrict__ outp) {
  __shared__ __align__(16) float    sc[16][Cdim + 4];
  __shared__ __align__(16) _Float16 ph[16][Cdim + 16];
  const int blk = blockIdx.x;
  const int qb = blk & 31;
  const int h  = (blk >> 5) & 7;
  const int b  = blk >> 8;
  const int lane = threadIdx.x;
  const int r = lane & 15;
  const int koff = (lane & 16) ? 8 : 0;
  const int half = lane >> 4;
  const int rs = 3 * Ddim;                                  // 1536
  const _Float16* Qb = qkv + (size_t)b * Cdim * rs + h * HD;
  const _Float16* Kb = Qb + Ddim;
  const _Float16* Vb = Qb + 2 * Ddim;

  // Q fragments for rows qb*16..+15, head-dim 0..63
  const _Float16* qp = Qb + (size_t)(qb * 16 + r) * rs + koff;
  v16h qa0 = load_frag(qp);
  v16h qa1 = load_frag(qp + 32);

  // scores = Q K^T / 8  -> LDS (f32)
  for (int kb = 0; kb < 32; ++kb) {
    const _Float16* kp = Kb + (size_t)(kb * 16 + r) * rs + koff;
    v16h b0 = load_frag(kp);
    v16h b1 = load_frag(kp + 32);
    v8f c = {};
    c = wmma32(qa0, b0, c);
    c = wmma32(qa1, b1, c);
#pragma unroll
    for (int j = 0; j < 8; ++j)
      sc[half * 8 + j][kb * 16 + r] = c[j] * 0.125f;
  }
  __syncthreads();
  // softmax: lane handles row r, column-half `half`; result -> ph (f16)
  {
    const int c0 = half * 256;
    float mx = -3.402823466e38f;
    for (int cc = 0; cc < 256; ++cc) mx = fmaxf(mx, sc[r][c0 + cc]);
    mx = fmaxf(mx, __shfl_xor(mx, 16));
    float sum = 0.f;
    for (int cc = 0; cc < 256; ++cc) {
      float e = __expf(sc[r][c0 + cc] - mx);
      sc[r][c0 + cc] = e;
      sum += e;
    }
    sum += __shfl_xor(sum, 16);
    float inv = 1.f / sum;
    for (int cc = 0; cc < 256; ++cc)
      ph[r][c0 + cc] = (_Float16)(sc[r][c0 + cc] * inv);
  }
  __syncthreads();
  // O = P * V  (16 x 64)
  v8f o[4] = {};
  for (int kb = 0; kb < 16; ++kb) {
    v16h pa = load_frag(&ph[r][kb * 32 + koff]);
#pragma unroll
    for (int dg = 0; dg < 4; ++dg) {
      v16h pb;
#pragma unroll
      for (int i = 0; i < 8; ++i) {
        pb[i]     = Vb[(size_t)(kb * 32 + koff + i) * rs + dg * 16 + r];
        pb[i + 8] = Vb[(size_t)(kb * 32 + 16 + koff + i) * rs + dg * 16 + r];
      }
      o[dg] = wmma32(pa, pb, o[dg]);
    }
  }
#pragma unroll
  for (int dg = 0; dg < 4; ++dg)
#pragma unroll
    for (int j = 0; j < 8; ++j) {
      int q = qb * 16 + half * 8 + j;
      outp[((size_t)b * Cdim + q) * Ddim + h * HD + dg * 16 + r] =
          (_Float16)o[dg][j];
    }
}

// ---------------------------------------------------------------------------
// LayerNorm( X + Res ): writes f32 Out and f16 OutH. D=512, 256 thr x 2 elem.
// ---------------------------------------------------------------------------
__global__ __launch_bounds__(256) void ln_res(const float* __restrict__ X,
                                              const float* __restrict__ Res,
                                              const float* __restrict__ w,
                                              const float* __restrict__ bb,
                                              float* __restrict__ Out,
                                              _Float16* __restrict__ OutH) {
  __shared__ float sh[16];
  const int row = blockIdx.x;
  const int t = threadIdx.x;
  const float* xp = X + (size_t)row * Ddim;
  const float* rp = Res + (size_t)row * Ddim;
  float v0 = xp[t] + rp[t];
  float v1 = xp[t + 256] + rp[t + 256];
  float s = v0 + v1, s2 = v0 * v0 + v1 * v1;
  for (int m = 16; m >= 1; m >>= 1) { s += __shfl_xor(s, m); s2 += __shfl_xor(s2, m); }
  if ((t & 31) == 0) { sh[t >> 5] = s; sh[8 + (t >> 5)] = s2; }
  __syncthreads();
  s = 0.f; s2 = 0.f;
  for (int wv = 0; wv < 8; ++wv) { s += sh[wv]; s2 += sh[8 + wv]; }
  float mu = s * (1.f / Ddim);
  float var = s2 * (1.f / Ddim) - mu * mu;
  float inv = rsqrtf(var + 1e-5f);
  float o0 = (v0 - mu) * inv * w[t] + bb[t];
  float o1 = (v1 - mu) * inv * w[t + 256] + bb[t + 256];
  Out[(size_t)row * Ddim + t]       = o0;
  Out[(size_t)row * Ddim + t + 256] = o1;
  OutH[(size_t)row * Ddim + t]       = (_Float16)o0;
  OutH[(size_t)row * Ddim + t + 256] = (_Float16)o1;
}

// ---------------------------------------------------------------------------
// Expand chunks to tokens + final LayerNorm (f32 out)
// ---------------------------------------------------------------------------
__global__ __launch_bounds__(256) void expand_ln(const float* __restrict__ X,
                                                 const int* __restrict__ seg,
                                                 const float* __restrict__ w,
                                                 const float* __restrict__ bb,
                                                 float* __restrict__ Out) {
  __shared__ float sh[16];
  const int row = blockIdx.x;               // b*S + s
  const int b = row >> 13;                  // / 8192
  const int t = threadIdx.x;
  const int c = seg[row];
  const float* xp = X + ((size_t)b * Cdim + c) * Ddim;
  float v0 = xp[t], v1 = xp[t + 256];
  float s = v0 + v1, s2 = v0 * v0 + v1 * v1;
  for (int m = 16; m >= 1; m >>= 1) { s += __shfl_xor(s, m); s2 += __shfl_xor(s2, m); }
  if ((t & 31) == 0) { sh[t >> 5] = s; sh[8 + (t >> 5)] = s2; }
  __syncthreads();
  s = 0.f; s2 = 0.f;
  for (int wv = 0; wv < 8; ++wv) { s += sh[wv]; s2 += sh[8 + wv]; }
  float mu = s * (1.f / Ddim);
  float var = s2 * (1.f / Ddim) - mu * mu;
  float inv = rsqrtf(var + 1e-5f);
  Out[(size_t)row * Ddim + t]       = (v0 - mu) * inv * w[t] + bb[t];
  Out[(size_t)row * Ddim + t + 256] = (v1 - mu) * inv * w[t + 256] + bb[t + 256];
}

// ---------------------------------------------------------------------------
extern "C" void kernel_launch(void* const* d_in, const int* in_sizes, int n_in,
                              void* d_out, int out_size, void* d_ws, size_t ws_size,
                              hipStream_t stream) {
  (void)in_sizes; (void)n_in; (void)out_size; (void)ws_size;
  const float* tokens = (const float*)d_in[0];
  const int*   seg    = (const int*)d_in[1];
  const float* qkv_w  = (const float*)d_in[2];
  const float* qkv_b  = (const float*)d_in[3];
  const float* out_w  = (const float*)d_in[4];
  const float* out_b  = (const float*)d_in[5];
  const float* ln1_w  = (const float*)d_in[6];
  const float* ln1_b  = (const float*)d_in[7];
  const float* ln2_w  = (const float*)d_in[8];
  const float* ln2_b  = (const float*)d_in[9];
  const float* ff1_w  = (const float*)d_in[10];
  const float* ff1_b  = (const float*)d_in[11];
  const float* ff2_w  = (const float*)d_in[12];
  const float* ff2_b  = (const float*)d_in[13];
  const float* fln_w  = (const float*)d_in[14];
  const float* fln_b  = (const float*)d_in[15];
  float* outp = (float*)d_out;

  const int M = Bdim * Cdim;                                // 4096
  char* ws = (char*)d_ws;
  float*    x     = (float*)ws;    ws += (size_t)M * Ddim * 4;
  float*    proj  = (float*)ws;    ws += (size_t)M * Ddim * 4;
  _Float16* xh    = (_Float16*)ws; ws += (size_t)M * Ddim * 2;
  _Float16* qkvh  = (_Float16*)ws; ws += (size_t)M * 3 * Ddim * 2;
  _Float16* attnh = (_Float16*)ws; ws += (size_t)M * Ddim * 2;
  _Float16* hbufh = (_Float16*)ws; ws += (size_t)M * DFFdim * 2;
  _Float16* qkvWh = (_Float16*)ws; ws += (size_t)Lnum * 3 * Ddim * Ddim * 2;
  _Float16* outWh = (_Float16*)ws; ws += (size_t)Lnum * Ddim * Ddim * 2;
  _Float16* ff1Wh = (_Float16*)ws; ws += (size_t)Lnum * DFFdim * Ddim * 2;
  _Float16* ff2Wh = (_Float16*)ws; ws += (size_t)Lnum * Ddim * DFFdim * 2;
  int*      bnds  = (int*)ws;

  auto cast_blocks = [](size_t n) { return (int)((n / 4 + 255) / 256); };
  const size_t nQkvW = (size_t)Lnum * 3 * Ddim * Ddim;
  const size_t nOutW = (size_t)Lnum * Ddim * Ddim;
  const size_t nFf1W = (size_t)Lnum * DFFdim * Ddim;
  const size_t nFf2W = (size_t)Lnum * Ddim * DFFdim;
  cast_f16<<<cast_blocks(nQkvW), 256, 0, stream>>>(qkv_w, qkvWh, (int)nQkvW);
  cast_f16<<<cast_blocks(nOutW), 256, 0, stream>>>(out_w, outWh, (int)nOutW);
  cast_f16<<<cast_blocks(nFf1W), 256, 0, stream>>>(ff1_w, ff1Wh, (int)nFf1W);
  cast_f16<<<cast_blocks(nFf2W), 256, 0, stream>>>(ff2_w, ff2Wh, (int)nFf2W);

  seg_bounds<<<(Bdim * Sdim) / 256, 256, 0, stream>>>(seg, bnds);
  seg_pool<<<M, 128, 0, stream>>>(tokens, bnds, x, xh);

  auto gemm_blocks = [](int m, int n) { return ((m / 32) * (n / 64)) / 8; };

  for (int l = 0; l < Lnum; ++l) {
    gemm_wmma<<<gemm_blocks(M, 3 * Ddim), 256, 0, stream>>>(
        xh, qkvWh + (size_t)l * 3 * Ddim * Ddim, qkv_b + (size_t)l * 3 * Ddim,
        qkvh, M, 3 * Ddim, Ddim, 0, 1);
    attn_wmma<<<Bdim * Hn * (Cdim / 16), 32, 0, stream>>>(qkvh, attnh);
    gemm_wmma<<<gemm_blocks(M, Ddim), 256, 0, stream>>>(
        attnh, outWh + (size_t)l * Ddim * Ddim, out_b + (size_t)l * Ddim,
        proj, M, Ddim, Ddim, 0, 0);
    ln_res<<<M, 256, 0, stream>>>(x, proj, ln1_w + (size_t)l * Ddim,
                                  ln1_b + (size_t)l * Ddim, x, xh);
    gemm_wmma<<<gemm_blocks(M, DFFdim), 256, 0, stream>>>(
        xh, ff1Wh + (size_t)l * DFFdim * Ddim, ff1_b + (size_t)l * DFFdim,
        hbufh, M, DFFdim, Ddim, 1, 1);
    gemm_wmma<<<gemm_blocks(M, Ddim), 256, 0, stream>>>(
        hbufh, ff2Wh + (size_t)l * Ddim * DFFdim, ff2_b + (size_t)l * Ddim,
        proj, M, Ddim, DFFdim, 0, 0);
    ln_res<<<M, 256, 0, stream>>>(x, proj, ln2_w + (size_t)l * Ddim,
                                  ln2_b + (size_t)l * Ddim, x, xh);
  }
  expand_ln<<<Bdim * Sdim, 256, 0, stream>>>(x, seg, fln_w, fln_b, outp);
}
